// ResidualVQ_46162308497955
// MI455X (gfx1250) — matrix operand order repaired and checked by reading, not compiled, not run on hardware
//
#include <hip/hip_runtime.h>

// ---------------------------------------------------------------------------
// Residual VQ fused kernels for gfx1250 (MI455X).
//   Q=4 quantizers, K=1024 codes, D=256 dims, M=B*S=16384 tokens.
//
// rvq_prep : codebooks f32 -> bf16 (into d_ws, L2-resident) + ||e||^2 norms.
// rvq_main : fused 4-quantizer loop. Distances via v_wmma_f32_16x16x32_bf16.
//            Residual lives in VGPRs in the WMMA A-fragment layout.
//            Codebook tiles arrive in LDS via double-buffered
//            global_load_async_to_lds_b128 (ASYNCcnt) overlapped with WMMA.
//            All 8 B-fragments are preloaded before the WMMA chain so the
//            scheduler can use partial s_wait_dscnt and stream the WMMAs.
// rvq_loss : deterministic fixed-order loss reduction.
//
// Outputs (concatenated float): quantized_output[16384*256],
//   indices[4*16384] (as float), losses[4].
// ---------------------------------------------------------------------------

typedef __bf16 bf16_t;
typedef __attribute__((ext_vector_type(16))) __bf16 v16bf;
typedef __attribute__((ext_vector_type(8)))  float  v8f;

#define QN        4
#define KCODES    1024
#define DIMS      256
#define MTOK      16384
#define NWAVE     4
#define TPB       (NWAVE * 32)       // 128 threads
#define TOKWG     (NWAVE * 16)       // 64 tokens per workgroup
#define NWG       (MTOK / TOKWG)     // 256 workgroups
#define NPART     (NWG * NWAVE)      // 1024 loss partials per quantizer
#define NTILES    (QN * KCODES / 16) // 256 total 16-code tiles
#define TILE_BYTES 8192              // 16 codes * 256 dims * 2B
#define OUT_IDX_OFF  (MTOK * DIMS)
#define OUT_LOSS_OFF (OUT_IDX_OFF + QN * MTOK)

// d_ws layout (bytes):
//   [0, 2MB)            bf16 codebooks  [4][1024][256]
//   [2MB, 2MB+16KB)     f32 norms       [4][1024]
//   [2MB+16KB, +16KB)   f32 loss partials [4][1024]
#define WS_CB16_ELEMS (QN * KCODES * DIMS)
#define WS_NORMS      (QN * KCODES)

__device__ __forceinline__ unsigned short f2bf_u(float f) {
  union { float f; unsigned u; } a; a.f = f;
  unsigned u = a.u;
  u += 0x7fffu + ((u >> 16) & 1u);            // round-to-nearest-even
  return (unsigned short)(u >> 16);
}

__device__ __forceinline__ bf16_t f2bf(float f) {
  union { unsigned short s; bf16_t b; } r;
  r.s = f2bf_u(f);
  return r.b;
}

__device__ __forceinline__ unsigned pack2(float lo, float hi) {
  return (unsigned)f2bf_u(lo) | ((unsigned)f2bf_u(hi) << 16);
}

// Issue 4 coalesced 512B async global->LDS copies (this wave's 2KB share of
// an 8KB tile). One ASYNCcnt increment per instruction, 4 per wave.
__device__ __forceinline__ void issue_tile_async(unsigned lds_addr,
                                                 unsigned long long gaddr) {
#pragma unroll
  for (int j = 0; j < 4; ++j) {
    asm volatile("global_load_async_to_lds_b128 %0, %1, off"
                 :
                 : "v"(lds_addr + (unsigned)(j * 512)),
                   "v"(gaddr + (unsigned long long)(j * 512))
                 : "memory");
  }
}

// Load the 16 dims this lane owns for chunk base b0:
//   dims [b0, b0+16) -> elems 0..15
#define LD16(dst, p, b0)                                                     \
  {                                                                          \
    float4 _a = *(const float4*)((p) + (b0));                                \
    float4 _b = *(const float4*)((p) + (b0) + 4);                            \
    float4 _c = *(const float4*)((p) + (b0) + 8);                            \
    float4 _d = *(const float4*)((p) + (b0) + 12);                           \
    dst[0] = _a.x;  dst[1] = _a.y;  dst[2]  = _a.z;  dst[3]  = _a.w;         \
    dst[4] = _b.x;  dst[5] = _b.y;  dst[6]  = _b.z;  dst[7]  = _b.w;         \
    dst[8] = _c.x;  dst[9] = _c.y;  dst[10] = _c.z;  dst[11] = _c.w;         \
    dst[12] = _d.x; dst[13] = _d.y; dst[14] = _d.z;  dst[15] = _d.w;         \
  }

// ---------------------------------------------------------------------------
// Prep: one wave per code. Convert 256 dims f32->bf16 and compute ||e||^2
// with a deterministic 32-lane butterfly. 512 blocks x 256 threads.
// ---------------------------------------------------------------------------
__global__ __launch_bounds__(256) void rvq_prep(const float* __restrict__ cb,
                                                bf16_t* __restrict__ cb16,
                                                float* __restrict__ norms) {
  const int wave = threadIdx.x >> 5;
  const int lane = threadIdx.x & 31;
  const int code = blockIdx.x * 8 + wave;        // 0 .. QN*KCODES-1
  const float* src = cb + (size_t)code * DIMS + lane * 8;

  float4 a = *(const float4*)(src);
  float4 b = *(const float4*)(src + 4);
  float n = 0.0f;
  n = fmaf(a.x, a.x, n); n = fmaf(a.y, a.y, n);
  n = fmaf(a.z, a.z, n); n = fmaf(a.w, a.w, n);
  n = fmaf(b.x, b.x, n); n = fmaf(b.y, b.y, n);
  n = fmaf(b.z, b.z, n); n = fmaf(b.w, b.w, n);
#pragma unroll
  for (int st = 1; st < 32; st <<= 1) n += __shfl_xor(n, st, 32);

  uint4 u;
  u.x = pack2(a.x, a.y); u.y = pack2(a.z, a.w);
  u.z = pack2(b.x, b.y); u.w = pack2(b.z, b.w);
  *(uint4*)(cb16 + (size_t)code * DIMS + lane * 8) = u;
  if (lane == 0) norms[code] = n;
}

// ---------------------------------------------------------------------------
// Main fused kernel.
// ---------------------------------------------------------------------------
__global__ __launch_bounds__(TPB) void rvq_main(const float* __restrict__ x,
                                                const float* __restrict__ cb,
                                                const bf16_t* __restrict__ cb16,
                                                const float* __restrict__ norms,
                                                float* __restrict__ out,
                                                float* __restrict__ lossp) {
  __shared__ __align__(32) bf16_t sB[2 * 16 * DIMS];  // double-buffered tile
  __shared__ int sIdx[TOKWG];

  const int tid   = threadIdx.x;
  const int wave  = tid >> 5;
  const int lane  = tid & 31;
  const int hi    = lane >> 4;            // 0: lanes 0-15, 1: lanes 16-31
  const int tloc  = lane & 15;            // this lane's token (within wave)
  const int off   = hi * 16;              // K sub-offset for A-fragment rows
  const int wg    = blockIdx.x;
  const int gtok0 = wg * TOKWG + wave * 16;

  const float* xr = x + (size_t)(gtok0 + tloc) * DIMS;

  // Residual in registers, A-fragment layout:
  // res[c][e] = residual[token tloc][ 32c + off + e ]
  float res[8][16];
#pragma unroll
  for (int c = 0; c < 8; ++c) {
    LD16(res[c], xr, 32 * c + off);
  }

  // Async-copy bases (generic LDS pointer low 32 bits == LDS byte offset).
  const unsigned sb_base = (unsigned)(size_t)(void*)sB;
  const unsigned long long cb16_base = (unsigned long long)(size_t)cb16;
  const unsigned lane_off = (unsigned)(wave * 2048 + lane * 16);

  // Prologue: start DMA of tile s=0 into buffer 0.
  issue_tile_async(sb_base + lane_off, cb16_base + lane_off);

  for (int q = 0; q < QN; ++q) {
    const float* cbq = cb + (size_t)q * KCODES * DIMS;
    const float* nq  = norms + (q << 10);

    // Convert this quantizer's residual to bf16 A fragments (once per q).
    v16bf abf[8];
#pragma unroll
    for (int c = 0; c < 8; ++c) {
#pragma unroll
      for (int e = 0; e < 16; ++e) abf[c][e] = f2bf(res[c][e]);
    }

    float bestv[8];
    int   besti[8];
#pragma unroll
    for (int r = 0; r < 8; ++r) { bestv[r] = 3.4e38f; besti[r] = 0; }

    for (int t = 0; t < 64; ++t) {
      const int s = (q << 6) + t;          // global tile stream index

      // Kick DMA for tile s+1 into the other buffer (safe: that buffer was
      // consumed at iteration t-1 and fenced by its trailing barrier), then
      // wait for our own tile-s copies (in-order completion => <=4 works).
      if (s + 1 < NTILES) {
        issue_tile_async(
            sb_base + (unsigned)(((s + 1) & 1) * TILE_BYTES) + lane_off,
            cb16_base + ((unsigned long long)(s + 1) * TILE_BYTES) + lane_off);
        asm volatile("s_wait_asynccnt 0x4" ::: "memory");
      } else {
        asm volatile("s_wait_asynccnt 0x0" ::: "memory");
      }
      __syncthreads();   // all waves' tile-s shares are in LDS

      const bf16_t* sbuf = sB + (t & 1) * (16 * DIMS);
      const float nrm = nq[(t << 4) + tloc];   // ||e||^2, L2-hot

      // ---- preload all 8 B fragments (16x ds_load_b128), then stream the
      //      8 chained WMMAs so DS latency is overlapped once, not 8x ----
      v16bf bfrag[8];
#pragma unroll
      for (int c = 0; c < 8; ++c) {
        // B fragment (32x16 bf16): lane-half hi selects K 0-15 / 16-31
        bfrag[c] = *(const v16bf*)(sbuf + (tloc * DIMS + 32 * c + hi * 16));
      }

      v8f acc = {0.f, 0.f, 0.f, 0.f, 0.f, 0.f, 0.f, 0.f};
#pragma unroll
      for (int c = 0; c < 8; ++c) {
        acc = __builtin_amdgcn_wmma_f32_16x16x32_bf16(
            false, abf[c], false, bfrag[c], (short)0, acc, false, false);
      }

      // score = ||e||^2 - 2 * dot  (||x||^2 constant per token: dropped)
      const int ni = (t << 4) + tloc;
#pragma unroll
      for (int r = 0; r < 8; ++r) {
        float sc = fmaf(-2.0f, acc[r], nrm);
        if (sc < bestv[r]) { bestv[r] = sc; besti[r] = ni; }
      }

      __syncthreads();   // everyone done reading this buffer before the
                         // iteration t+1 DMA refills it
    }

    // ---- final argmin across the 16 code-columns (lanes) per token ----
#pragma unroll
    for (int r = 0; r < 8; ++r) {
#pragma unroll
      for (int st = 1; st < 16; st <<= 1) {
        float ov = __shfl_xor(bestv[r], st, 32);
        int   oi = __shfl_xor(besti[r], st, 32);
        bool take = (ov < bestv[r]) || ((ov == bestv[r]) && (oi < besti[r]));
        if (take) { bestv[r] = ov; besti[r] = oi; }
      }
    }
    // lanes 0-15 hold tokens 0..7 in slots r; lanes 16-31 hold tokens 8..15
    if (lane == 0) {
#pragma unroll
      for (int r = 0; r < 8; ++r) {
        sIdx[wave * 16 + r] = besti[r];
        out[OUT_IDX_OFF + q * MTOK + gtok0 + r] = (float)besti[r];
      }
    }
    if (lane == 16) {
#pragma unroll
      for (int r = 0; r < 8; ++r) {
        sIdx[wave * 16 + 8 + r] = besti[r];
        out[OUT_IDX_OFF + q * MTOK + gtok0 + 8 + r] = (float)besti[r];
      }
    }
    __syncthreads();

    // ---- gather chosen code (f32), update residual, commitment loss ----
    const int myidx = sIdx[wave * 16 + tloc];
    const float* ev = cbq + (size_t)myidx * DIMS;
    float lacc = 0.0f;
#pragma unroll
    for (int c = 0; c < 8; ++c) {
      float e16[16];
      LD16(e16, ev, 32 * c + off);
#pragma unroll
      for (int e = 0; e < 16; ++e) {
        float nr = res[c][e] - e16[e];   // residual - quantize
        res[c][e] = nr;
        lacc = fmaf(nr, nr, lacc);       // (q - residual)^2
      }
    }
    // deterministic full-wave butterfly sum
#pragma unroll
    for (int st = 1; st < 32; st <<= 1) lacc += __shfl_xor(lacc, st, 32);
    if (lane == 0) lossp[q * NPART + wg * NWAVE + wave] = lacc;
  }

  // quantized_output = x - residual_final  (straight-through forward value)
  float* op = out + (size_t)(gtok0 + tloc) * DIMS;
#pragma unroll
  for (int c = 0; c < 8; ++c) {
    const int b0 = 32 * c + off;
    float xv[16];
    LD16(xv, xr, b0);
    *(float4*)(op + b0) =
        make_float4(xv[0] - res[c][0], xv[1] - res[c][1],
                    xv[2] - res[c][2], xv[3] - res[c][3]);
    *(float4*)(op + b0 + 4) =
        make_float4(xv[4] - res[c][4], xv[5] - res[c][5],
                    xv[6] - res[c][6], xv[7] - res[c][7]);
    *(float4*)(op + b0 + 8) =
        make_float4(xv[8] - res[c][8], xv[9] - res[c][9],
                    xv[10] - res[c][10], xv[11] - res[c][11]);
    *(float4*)(op + b0 + 12) =
        make_float4(xv[12] - res[c][12], xv[13] - res[c][13],
                    xv[14] - res[c][14], xv[15] - res[c][15]);
  }
}

// Deterministic fixed-order loss reduction: losses[q] = mean over M*D.
__global__ __launch_bounds__(32) void rvq_loss(const float* __restrict__ lossp,
                                               float* __restrict__ out) {
  const int t = threadIdx.x;
  if (t < QN) {
    float s = 0.0f;
    for (int i = 0; i < NPART; ++i) s += lossp[t * NPART + i];
    out[OUT_LOSS_OFF + t] = s * (1.0f / ((float)MTOK * (float)DIMS));
  }
}

extern "C" void kernel_launch(void* const* d_in, const int* in_sizes, int n_in,
                              void* d_out, int out_size, void* d_ws,
                              size_t ws_size, hipStream_t stream) {
  (void)in_sizes; (void)n_in; (void)out_size; (void)ws_size;
  const float* x  = (const float*)d_in[0];   // [8,2048,256] f32
  const float* cb = (const float*)d_in[1];   // [4,1024,256] f32
  float* out = (float*)d_out;

  bf16_t* cb16  = (bf16_t*)d_ws;                       // 2 MB
  float*  norms = (float*)((char*)d_ws + WS_CB16_ELEMS * sizeof(bf16_t));
  float*  lossp = norms + WS_NORMS;

  rvq_prep<<<(QN * KCODES) / 8, 256, 0, stream>>>(cb, cb16, norms);
  rvq_main<<<NWG, TPB, 0, stream>>>(x, cb, cb16, norms, out, lossp);
  rvq_loss<<<1, 32, 0, stream>>>(lossp, out);
}
